// CausalSelfAttentionBottleneck_7361573946034
// MI455X (gfx1250) — matrix operand order repaired and checked by reading, not compiled, not run on hardware
//
#include <hip/hip_runtime.h>
#include <math.h>

// ---------------------------------------------------------------------------
// CDNA5 (gfx1250) causal self-attention bottleneck, bf16 WMMA pipeline.
//   B=4, T=2048, C=1024, H=16, D=64
// Stages:
//   1) f32->bf16 convert (x) and convert+transpose (weights)
//   2) TDM-staged WMMA GEMM for Q/K/V projections (V written transposed
//      per head so attention needs no transposes at all)
//   3) barrier-free flash attention (WMMA QK^T and PV, online softmax,
//      learned null-KV column)
//   4) TDM-staged WMMA GEMM for the output projection (f32 out)
// ---------------------------------------------------------------------------

typedef __attribute__((ext_vector_type(16))) __bf16 v16bf;
typedef __attribute__((ext_vector_type(8)))  float  v8f;

// Tensor-DMA descriptor vector types (clang-23 6-arg builtin signature).
typedef __attribute__((ext_vector_type(4))) unsigned int tdm4u;
typedef __attribute__((ext_vector_type(8))) int          tdm8i;
typedef __attribute__((ext_vector_type(4))) int          tdm4i;

struct alignas(16) U4 { unsigned int x, y, z, w; };
struct alignas(8)  U2 { unsigned int x, y; };

union Frag16 { v16bf v; U4 q[2]; };

__device__ __forceinline__ unsigned short f2bf(float x) {
  unsigned int u = __float_as_uint(x);
  u += 0x7fffu + ((u >> 16) & 1u);          // round-to-nearest-even
  return (unsigned short)(u >> 16);
}
__device__ __forceinline__ float bf2f(unsigned short b) {
  return __uint_as_float(((unsigned int)b) << 16);
}

// Load a 16-element bf16 fragment from two 16B-aligned 8-element chunks.
__device__ __forceinline__ v16bf load_frag(const unsigned short* lo,
                                           const unsigned short* hi) {
  Frag16 f;
  f.q[0] = *(const U4*)lo;
  f.q[1] = *(const U4*)hi;
  return f.v;
}
// Load a 16-element bf16 fragment from 32 contiguous bytes (16B aligned).
__device__ __forceinline__ v16bf load_frag32(const unsigned short* p) {
  Frag16 f;
  f.q[0] = *(const U4*)p;
  f.q[1] = *(const U4*)(p + 8);
  return f.v;
}

__device__ __forceinline__ v8f wmma_bf16(v16bf a, v16bf b, v8f c) {
  // (neg_a, A, neg_b, B, c_mod, C, reuse_a, reuse_b)
  return __builtin_amdgcn_wmma_f32_16x16x32_bf16(false, a, false, b,
                                                 (short)0, c, false, false);
}

// Wave-local LDS ordering: DS ops of one wave execute in order; this just
// drains DScnt and stops the compiler reordering across the round trip.
__device__ __forceinline__ void lds_fence_wave() {
  asm volatile("s_wait_dscnt 0x0" ::: "memory");
}

// ---------------------------------------------------------------------------
// TDM: DMA a 2D bf16 tile [tile_d1 rows x tile_d0 elems] from a row-major
// tensor (row stride = stride0 elems) into LDS with per-row padding
// (pad codes per D# spec: interval 4 -> every 32 DWORDs, amount 3 -> 4 DWORDs).
// ---------------------------------------------------------------------------
__device__ __forceinline__ void tdm_load_2d(unsigned lds_addr,
                                            unsigned long long gaddr,
                                            unsigned tensor_d0, unsigned tensor_d1,
                                            unsigned tile_d0, unsigned tile_d1,
                                            unsigned stride0,
                                            unsigned pad_int, unsigned pad_amt) {
  tdm4u g0;
  g0[0] = 1u;                                       // count=1, user descriptor
  g0[1] = lds_addr;                                 // LDS byte address
  g0[2] = (unsigned)gaddr;                          // global addr [31:0]
  g0[3] = (unsigned)(gaddr >> 32) | (2u << 30);     // addr[56:32] | type=2
  tdm8i g1;
  g1[0] = (int)((1u << 16) |                        // data_size = 2 bytes
                (1u << 20) |                        // pad_enable
                (pad_int << 22) | (pad_amt << 25)); // LDS row padding
  g1[1] = (int)((tensor_d0 & 0xFFFFu) << 16);       // tensor_dim0 lo16
  g1[2] = (int)((tensor_d0 >> 16) | ((tensor_d1 & 0xFFFFu) << 16));
  g1[3] = (int)((tensor_d1 >> 16) | (tile_d0 << 16));
  g1[4] = (int)tile_d1;                             // tile_dim1 (tile_dim2=0)
  g1[5] = (int)stride0;                             // tensor_dim0_stride lo32
  g1[6] = 0;
  g1[7] = 0;
  tdm4i z4 = {0, 0, 0, 0};                          // 2D: groups 2/3 unused
  tdm8i z8 = {0, 0, 0, 0, 0, 0, 0, 0};
  __builtin_amdgcn_tensor_load_to_lds(g0, g1, z4, z4, z8, 0);
}

// ---------------------------------------------------------------------------
// f32 -> bf16 conversion
// ---------------------------------------------------------------------------
__global__ void cvt_f32_bf16(const float* __restrict__ src,
                             unsigned short* __restrict__ dst, int n) {
  int i = (blockIdx.x * blockDim.x + threadIdx.x) * 4;
  if (i + 3 < n) {
    float a = src[i], b = src[i + 1], c = src[i + 2], d = src[i + 3];
    U2 pk;
    pk.x = (unsigned)f2bf(a) | ((unsigned)f2bf(b) << 16);
    pk.y = (unsigned)f2bf(c) | ((unsigned)f2bf(d) << 16);
    *(U2*)(dst + i) = pk;
  } else {
    for (; i < n; ++i) dst[i] = f2bf(src[i]);
  }
}

// ---------------------------------------------------------------------------
// f32 [R x C] -> bf16 transposed [C x R] (weights, done once; tiny cost)
// block (32,8), grid (C/32, R/32)
// ---------------------------------------------------------------------------
__global__ void cvt_transpose_bf16(const float* __restrict__ src,
                                   unsigned short* __restrict__ dst,
                                   int R, int C) {
  __shared__ unsigned short t[32][33];
  const int c0 = blockIdx.x * 32, r0 = blockIdx.y * 32;
  const int x = threadIdx.x, y = threadIdx.y;
#pragma unroll
  for (int i = 0; i < 32; i += 8)
    t[y + i][x] = f2bf(src[(size_t)(r0 + y + i) * C + c0 + x]);
  __syncthreads();
#pragma unroll
  for (int i = 0; i < 32; i += 8)
    dst[(size_t)(c0 + y + i) * R + r0 + x] = t[x][y + i];
}

// ---------------------------------------------------------------------------
// bf16 GEMM: C[M,N] = A[M,K] * BT[N,K]^T   (B supplied pre-transposed)
// 128x128 tile / 256-thread block (8 waves, each 32x64), K-step 64.
// A and BT tiles are DMA'd into padded LDS by the Tensor Data Mover.
// MODE: 0 = bf16 row-major out, 1 = f32 row-major out,
//       2 = bf16 out transposed per head: [B,H,D,T] (for V)
// ---------------------------------------------------------------------------
template <int MODE>
__global__ __launch_bounds__(256) void gemm_bf16(
    const unsigned short* __restrict__ A, const unsigned short* __restrict__ BT,
    void* __restrict__ Cout, int M, int N, int K) {
  constexpr int LDA = 72;  // 64 elems + 8 pad = 144B rows (16B aligned)
  __shared__ __align__(16) unsigned short As[128 * LDA];  // [m][k]
  __shared__ __align__(16) unsigned short Bs[128 * LDA];  // [n][k]

  const int tid = threadIdx.x;
  const int wave = tid >> 5, lane = tid & 31;
  const int l15 = lane & 15;
  const int hsel = (lane < 16) ? 0 : 1;
  const int m0 = blockIdx.y * 128, n0 = blockIdx.x * 128;
  const int wm = (wave >> 1) * 32;  // wave rows within tile
  const int wn = (wave & 1) * 64;   // wave cols within tile

  const unsigned ldsA = (unsigned)(uintptr_t)&As[0];
  const unsigned ldsB = (unsigned)(uintptr_t)&Bs[0];

  v8f acc[2][4] = {};

  for (int k0 = 0; k0 < K; k0 += 64) {
    __syncthreads();  // previous tile fully consumed
    if (wave == 0) {
      tdm_load_2d(ldsA, (unsigned long long)(uintptr_t)(A + (size_t)m0 * K + k0),
                  (unsigned)K, (unsigned)M, 64u, 128u, (unsigned)K, 4u, 3u);
      __builtin_amdgcn_s_wait_tensorcnt(0);
    } else if (wave == 1) {
      tdm_load_2d(ldsB, (unsigned long long)(uintptr_t)(BT + (size_t)n0 * K + k0),
                  (unsigned)K, (unsigned)N, 64u, 128u, (unsigned)K, 4u, 3u);
      __builtin_amdgcn_s_wait_tensorcnt(0);
    }
    __syncthreads();  // tiles resident in LDS

#pragma unroll
    for (int kk = 0; kk < 2; ++kk) {  // two K=32 WMMA slabs
      v16bf af[2], bf[4];
#pragma unroll
      for (int i = 0; i < 2; ++i) {
        // A 16x32 frag, this code's contiguous k order (B matches): lane<16 -> K 0..15 ; lane>=16 -> K 16..31
        const unsigned short* p = As + (wm + i * 16 + l15) * LDA + kk * 32 + hsel * 16;
        af[i] = load_frag32(p);
      }
#pragma unroll
      for (int j = 0; j < 4; ++j) {
        // B 32x16 frag, same contiguous k order as A: lanes 0-15 -> K 0..15 at N=lane ; lanes 16-31 -> 16..31
        const unsigned short* p = Bs + (wn + j * 16 + l15) * LDA + kk * 32 + hsel * 16;
        bf[j] = load_frag32(p);
      }
#pragma unroll
      for (int i = 0; i < 2; ++i)
#pragma unroll
        for (int j = 0; j < 4; ++j)
          acc[i][j] = wmma_bf16(af[i], bf[j], acc[i][j]);
    }
  }

  // ---- epilogue: C layout VGPR r -> row r (lanes<16) / r+8 (lanes>=16)
  const int row_base = m0 + wm + (hsel ? 8 : 0);
#pragma unroll
  for (int i = 0; i < 2; ++i) {
#pragma unroll
    for (int r = 0; r < 8; ++r) {
      int row = row_base + i * 16 + r;
#pragma unroll
      for (int j = 0; j < 4; ++j) {
        int col = n0 + wn + j * 16 + l15;
        float v = acc[i][j][r];
        if (MODE == 1) {
          ((float*)Cout)[(size_t)row * N + col] = v;
        } else if (MODE == 0) {
          ((unsigned short*)Cout)[(size_t)row * N + col] = f2bf(v);
        } else {  // MODE 2: V transposed per head -> [B,H,D,T], T=2048, H=16
          int bb = row >> 11, t = row & 2047;
          int hh = col >> 6, dd = col & 63;
          ((unsigned short*)Cout)[(((size_t)bb * 16 + hh) * 64 + dd) * 2048 + t] =
              f2bf(v);
        }
      }
    }
  }
}

// ---------------------------------------------------------------------------
// Barrier-free flash attention with learned null-KV column.
// Grid: (T/64, H, B). 128 threads = 4 independent waves; each wave owns 16
// query rows. K streamed d-contiguous from row-major K; V streamed
// s-contiguous from the pre-transposed [B,H,D,T] layout. Only LDS use is the
// per-wave P C-layout -> A-layout round trip (wave-local ordering).
// ---------------------------------------------------------------------------
__global__ __launch_bounds__(128) void attn_flash(
    const unsigned short* __restrict__ Q, const unsigned short* __restrict__ K,
    const unsigned short* __restrict__ VT, unsigned short* __restrict__ Y,
    const float* __restrict__ null_k, const float* __restrict__ null_v,
    const float* __restrict__ logit_scale, int T) {
  constexpr int HD = 1024, Dh = 64;
  constexpr int LDP = 40;  // 80B padded rows (16B aligned)
  __shared__ __align__(16) unsigned short Pt[4][16 * LDP];

  const int tid = threadIdx.x, wave = tid >> 5, lane = tid & 31;
  const int l15 = lane & 15;
  const int hsel = (lane < 16) ? 0 : 1;
  const int h = blockIdx.y, b = blockIdx.z;
  const int qbase = blockIdx.x * 64;
  const float scale = __expf(logit_scale[h]) * 0.125f;  // exp(ls)/sqrt(64)
  const size_t rowoff = (size_t)b * T * HD + (size_t)h * Dh;
  const unsigned short* vbase = VT + ((size_t)(b * 16 + h) * Dh) * T;  // [64][T]

  // Q A-fragments (16 rows x 64 d -> two 16x32 fragments), held in registers
  const int row_min = qbase + wave * 16;
  const int qrow = row_min + l15;
  const unsigned short* qp = Q + rowoff + (size_t)qrow * HD;
  v16bf qf[2];
  qf[0] = load_frag32(qp + hsel * 16);
  qf[1] = load_frag32(qp + 32 + hsel * 16);

  v8f acc[4] = {};
  float mrun[8], lrun[8];
#pragma unroll
  for (int j = 0; j < 8; ++j) { mrun[j] = -1e30f; lrun[j] = 0.0f; }

  const int rbase = row_min + (hsel ? 8 : 0);
  const int s_end = row_min + 16;  // this wave's exact causal horizon

  unsigned short* pw = Pt[wave];

  for (int s0 = 0; s0 < s_end; s0 += 32) {
    // ---- S = Q K^T for 32 keys (two 16x16 tiles), K streamed from global
    v8f sv[2];
#pragma unroll
    for (int st = 0; st < 2; ++st) {
      const unsigned short* kp = K + rowoff + (size_t)(s0 + st * 16 + l15) * HD;
      v16bf kb0 = load_frag32(kp + hsel * 16);       // d 0..31
      v16bf kb1 = load_frag32(kp + 32 + hsel * 16);  // d 32..63
      v8f c = {};
      c = wmma_bf16(qf[0], kb0, c);
      c = wmma_bf16(qf[1], kb1, c);
      sv[st] = c;
    }

    // ---- online softmax (rows split across lane halves per C layout)
    const bool need_mask = (s0 + 31 > row_min);
    float p0[8], p1[8], alpha[8];
#pragma unroll
    for (int j = 0; j < 8; ++j) {
      float v0 = sv[0][j] * scale;
      float v1 = sv[1][j] * scale;
      if (need_mask) {
        int row = rbase + j;
        if (s0 + l15 > row) v0 = -1e30f;
        if (s0 + 16 + l15 > row) v1 = -1e30f;
      }
      float mx = fmaxf(v0, v1);
#pragma unroll
      for (int d = 1; d < 16; d <<= 1) mx = fmaxf(mx, __shfl_xor(mx, d, 32));
      float mnew = fmaxf(mrun[j], mx);
      float a = __expf(mrun[j] - mnew);
      p0[j] = __expf(v0 - mnew);
      p1[j] = __expf(v1 - mnew);
      float rs = p0[j] + p1[j];
#pragma unroll
      for (int d = 1; d < 16; d <<= 1) rs += __shfl_xor(rs, d, 32);
      lrun[j] = lrun[j] * a + rs;
      mrun[j] = mnew;
      alpha[j] = a;
    }
#pragma unroll
    for (int nb = 0; nb < 4; ++nb)
#pragma unroll
      for (int j = 0; j < 8; ++j) acc[nb][j] *= alpha[j];

    // ---- C-layout -> A-layout for P via wave-local LDS round trip
#pragma unroll
    for (int j = 0; j < 8; ++j) {
      int r = j + (hsel ? 8 : 0);
      pw[r * LDP + l15] = f2bf(p0[j]);
      pw[r * LDP + 16 + l15] = f2bf(p1[j]);
    }
    lds_fence_wave();  // wave-private tile; DS ops of a wave are in order
    const unsigned short* pr = pw + l15 * LDP + hsel * 16;
    v16bf pf = load_frag32(pr);

    // ---- O += P V : B fragments stream s-contiguous from transposed V
#pragma unroll
    for (int nb = 0; nb < 4; ++nb) {
      const unsigned short* vp =
          vbase + (size_t)(nb * 16 + l15) * T + s0 + hsel * 16;
      v16bf vb = load_frag32(vp);
      acc[nb] = wmma_bf16(pf, vb, acc[nb]);
    }
  }

  // ---- learned null-KV column (scalar epilogue)
  {
    const float* nk = null_k + h * Dh;
    const float* nv = null_v + h * Dh;
    const unsigned short* qd = Q + rowoff + (size_t)qrow * HD + hsel * 32;
    float part = 0.0f;
#pragma unroll
    for (int d = 0; d < 32; ++d) part += bf2f(qd[d]) * nk[hsel * 32 + d];
    part += __shfl_xor(part, 16, 32);
    float ln_lane = part * scale;  // null logit for row qrow
    float nvl[4];
#pragma unroll
    for (int nb = 0; nb < 4; ++nb) nvl[nb] = nv[nb * 16 + l15];
#pragma unroll
    for (int j = 0; j < 8; ++j) {
      float ln = __shfl(ln_lane, j + (hsel ? 8 : 0), 32);
      float mnew = fmaxf(mrun[j], ln);
      float a = __expf(mrun[j] - mnew);
      float p = __expf(ln - mnew);
      lrun[j] = lrun[j] * a + p;
      mrun[j] = mnew;
#pragma unroll
      for (int nb = 0; nb < 4; ++nb) acc[nb][j] = acc[nb][j] * a + p * nvl[nb];
    }
  }

  // ---- normalize and store y (bf16, [B*T, H*64] row-major)
#pragma unroll
  for (int j = 0; j < 8; ++j) {
    int row = rbase + j;
    float inv = 1.0f / lrun[j];
#pragma unroll
    for (int nb = 0; nb < 4; ++nb) {
      float v = acc[nb][j] * inv;
      Y[rowoff + (size_t)row * HD + nb * 16 + l15] = f2bf(v);
    }
  }
}

// ---------------------------------------------------------------------------
// Host-side launch
// ---------------------------------------------------------------------------
extern "C" void kernel_launch(void* const* d_in, const int* in_sizes, int n_in,
                              void* d_out, int out_size, void* d_ws,
                              size_t ws_size, hipStream_t stream) {
  (void)in_sizes; (void)n_in; (void)out_size; (void)ws_size;
  const float* x  = (const float*)d_in[0];
  const float* Wq = (const float*)d_in[1];
  const float* Wk = (const float*)d_in[2];
  const float* Wv = (const float*)d_in[3];
  const float* Wo = (const float*)d_in[4];
  const float* null_k = (const float*)d_in[5];
  const float* null_v = (const float*)d_in[6];
  const float* logit_scale = (const float*)d_in[7];
  float* out = (float*)d_out;

  const int Bn = 4, T = 2048, C = 1024, HD = 1024;
  const int M = Bn * T;  // 8192
  const size_t MC = (size_t)M * C, CC = (size_t)C * HD;

  unsigned short* ws   = (unsigned short*)d_ws;
  unsigned short* xb   = ws;             // [M, C]
  unsigned short* wqT  = xb + MC;        // [HD, C]  (transposed)
  unsigned short* wkT  = wqT + CC;
  unsigned short* wvT  = wkT + CC;
  unsigned short* woT  = wvT + CC;       // [C, HD]  (transposed)
  unsigned short* Qb   = woT + CC;       // [M, HD]
  unsigned short* Kb   = Qb + MC;        // [M, HD]
  unsigned short* VTb  = Kb + MC;        // [B, H, 64, T]
  unsigned short* Yb   = VTb + MC;       // [M, HD]   (total ~88 MB)

  // 1) converts: x plain; weights convert+transpose (all square 1024x1024)
  {
    int n = (int)MC;
    cvt_f32_bf16<<<(n + 1023) / 1024, 256, 0, stream>>>(x, xb, n);
    dim3 tb(32, 8), tg(HD / 32, C / 32);
    cvt_transpose_bf16<<<tg, tb, 0, stream>>>(Wq, wqT, C, HD);
    cvt_transpose_bf16<<<tg, tb, 0, stream>>>(Wk, wkT, C, HD);
    cvt_transpose_bf16<<<tg, tb, 0, stream>>>(Wv, wvT, C, HD);
    cvt_transpose_bf16<<<tg, tb, 0, stream>>>(Wo, woT, HD, C);
  }

  // 2) Q/K/V projections (TDM-staged WMMA GEMM); V written per-head transposed
  dim3 ggrid(HD / 128, M / 128);  // (8, 64)
  gemm_bf16<0><<<ggrid, 256, 0, stream>>>(xb, wqT, Qb, M, HD, C);
  gemm_bf16<0><<<ggrid, 256, 0, stream>>>(xb, wkT, Kb, M, HD, C);
  gemm_bf16<2><<<ggrid, 256, 0, stream>>>(xb, wvT, VTb, M, HD, C);

  // 3) Flash attention with null-KV column (barrier-free, WMMA)
  dim3 agrid(T / 64, 16, Bn);  // (32, 16, 4)
  attn_flash<<<agrid, 128, 0, stream>>>(Qb, Kb, VTb, Yb, null_k, null_v,
                                        logit_scale, T);

  // 4) Output projection: y @ Wo -> f32 d_out
  gemm_bf16<1><<<ggrid, 256, 0, stream>>>(Yb, woT, out, M, C, HD);
}
